// TransformerBlock_47141561041149
// MI455X (gfx1250) — matrix operand order repaired and checked by reading, not compiled, not run on hardware
//
#include <hip/hip_runtime.h>
#include <hip/hip_bf16.h>

typedef __attribute__((ext_vector_type(16))) __bf16 v16bf;
typedef __attribute__((ext_vector_type(8)))  __bf16 v8bf;
typedef __attribute__((ext_vector_type(8)))  float  v8f;

#define DIM_   1024
#define SEQ_   4096
#define NHEADS 16
#define HDIM   64
#define HIDDEN_ 4096

#define WMMA_BF16(a, b, c) \
  __builtin_amdgcn_wmma_f32_16x16x32_bf16(false, (a), false, (b), (short)0, (c), false, false)

// ---------- fragment helpers (contiguous k order, same for A and B) ----------
static __device__ __forceinline__ v16bf make_afrag(v8bf c0, v8bf c1) {
  return __builtin_shufflevector(c0, c1, 0,1,2,3,4,5,6,7,8,9,10,11,12,13,14,15);
}
// A: 16x32 bf16. lane: M = m0+(lane&15), K = (lane>>4)*16 + 0..15 (contiguous)
static __device__ __forceinline__ v16bf load_a_frag(const __bf16* A, int lda, int m0, int k0, int lane) {
  int l16 = lane & 15, half = lane >> 4;
  const __bf16* p = A + (size_t)(m0 + l16) * lda + k0 + half * 16;
  return *(const v16bf*)p;
}
// B: 32x16 bf16 = W^T. lane: N = n0+(lane&15), K = (lane>>4)*16 + 0..15 (contiguous in W row)
static __device__ __forceinline__ v16bf load_b_frag(const __bf16* W, int ldw, int n0, int k0, int lane) {
  int l16 = lane & 15, half = lane >> 4;
  return *(const v16bf*)(W + (size_t)(n0 + l16) * ldw + k0 + half * 16);
}

static __device__ __forceinline__ float red_max16(float v) {
  v = fmaxf(v, __shfl_xor(v, 1, 32));
  v = fmaxf(v, __shfl_xor(v, 2, 32));
  v = fmaxf(v, __shfl_xor(v, 4, 32));
  v = fmaxf(v, __shfl_xor(v, 8, 32));
  return v;
}
static __device__ __forceinline__ float red_sum16(float v) {
  v += __shfl_xor(v, 1, 32);
  v += __shfl_xor(v, 2, 32);
  v += __shfl_xor(v, 4, 32);
  v += __shfl_xor(v, 8, 32);
  return v;
}

// ---------- elementwise kernels ----------
__global__ void f32_to_bf16_kernel(const float* __restrict__ s, __bf16* __restrict__ d, int n) {
  int i = blockIdx.x * blockDim.x + threadIdx.x;
  if (i < n) d[i] = (__bf16)s[i];
}

__global__ __launch_bounds__(256) void rmsnorm_bf16_kernel(const float* __restrict__ x,
                                                           const float* __restrict__ g,
                                                           __bf16* __restrict__ out, int ncols) {
  __shared__ float red[32];
  int row = blockIdx.x;
  const float* xr = x + (size_t)row * ncols;
  float ss = 0.f;
  for (int c = threadIdx.x; c < ncols; c += blockDim.x) { float v = xr[c]; ss += v * v; }
  for (int s = 16; s > 0; s >>= 1) ss += __shfl_xor(ss, s, 32);
  int wid = threadIdx.x >> 5, ln = threadIdx.x & 31;
  if (ln == 0) red[wid] = ss;
  __syncthreads();
  if (wid == 0) {
    float v = (ln < (int)(blockDim.x >> 5)) ? red[ln] : 0.f;
    for (int s = 16; s > 0; s >>= 1) v += __shfl_xor(v, s, 32);
    if (ln == 0) red[0] = v;
  }
  __syncthreads();
  float inv = rsqrtf(red[0] / (float)ncols + 1e-6f);
  for (int c = threadIdx.x; c < ncols; c += blockDim.x)
    out[(size_t)row * ncols + c] = (__bf16)(xr[c] * inv * g[c]);
}

__global__ void rope_bf16_kernel(const float* __restrict__ qf, const float* __restrict__ kf,
                                 __bf16* __restrict__ qb, __bf16* __restrict__ kb, int seq) {
  int idx = blockIdx.x * blockDim.x + threadIdx.x;  // seq * NHEADS * (HDIM/2)
  if (idx >= seq * NHEADS * (HDIM / 2)) return;
  int pair = idx & 31;
  int t = idx >> 5;
  int h = t & (NHEADS - 1);
  int s = t >> 4;
  float inv = powf(10000.f, -(2.f * (float)pair) / (float)HDIM);
  float ang = (float)s * inv;
  float c = cosf(ang), sn = sinf(ang);
  size_t base = (size_t)s * DIM_ + h * HDIM + 2 * pair;
  float re = qf[base], im = qf[base + 1];
  qb[base]     = (__bf16)(re * c - im * sn);
  qb[base + 1] = (__bf16)(re * sn + im * c);
  re = kf[base]; im = kf[base + 1];
  kb[base]     = (__bf16)(re * c - im * sn);
  kb[base + 1] = (__bf16)(re * sn + im * c);
}

__global__ void silu_mul_bf16_kernel(__bf16* __restrict__ a, const __bf16* __restrict__ b, int n) {
  int i = blockIdx.x * blockDim.x + threadIdx.x;
  if (i < n) {
    float x = (float)a[i], y = (float)b[i];
    a[i] = (__bf16)((x / (1.f + __expf(-x))) * y);
  }
}

// ---------- WMMA GEMM: C[M,N] = A[M,K] * W[N,K]^T  (bf16 in, f32 accum) ----------
// STORE: 0 = f32 row-major (+optional residual), 1 = bf16 row-major, 2 = bf16 transposed [N,M]
template <int STORE>
__global__ __launch_bounds__(256) void gemm_wmma_kernel(const __bf16* __restrict__ A,
                                                        const __bf16* __restrict__ W,
                                                        void* __restrict__ Cv,
                                                        const float* __restrict__ resid,
                                                        int M, int N, int K) {
  int lane = threadIdx.x & 31;
  int wave = threadIdx.x >> 5;
  int wm = wave & 1, wn = wave >> 1;          // 2 x 4 waves
  int m0 = blockIdx.y * 64 + wm * 32;         // wave tile: 32 (M) x 64 (N)
  int n0 = blockIdx.x * 256 + wn * 64;

  v8f acc[2][4] = {};
  for (int k0 = 0; k0 < K; k0 += 32) {
    v16bf a0 = load_a_frag(A, K, m0, k0, lane);
    v16bf a1 = load_a_frag(A, K, m0 + 16, k0, lane);
    v16bf b[4];
#pragma unroll
    for (int j = 0; j < 4; j++) b[j] = load_b_frag(W, K, n0 + 16 * j, k0, lane);
#pragma unroll
    for (int j = 0; j < 4; j++) {
      acc[0][j] = WMMA_BF16(a0, b[j], acc[0][j]);
      acc[1][j] = WMMA_BF16(a1, b[j], acc[1][j]);
    }
  }

  int l16 = lane & 15, half = lane >> 4;
#pragma unroll
  for (int i = 0; i < 2; i++)
#pragma unroll
    for (int j = 0; j < 4; j++)
#pragma unroll
      for (int r = 0; r < 8; r++) {
        int row = m0 + 16 * i + half * 8 + r;
        int col = n0 + 16 * j + l16;
        float v = acc[i][j][r];
        if (STORE == 0) {
          if (resid) v += resid[(size_t)row * N + col];
          ((float*)Cv)[(size_t)row * N + col] = v;
        } else if (STORE == 1) {
          ((__bf16*)Cv)[(size_t)row * N + col] = (__bf16)v;
        } else {
          ((__bf16*)Cv)[(size_t)col * M + row] = (__bf16)v;
        }
      }
}

// ---------- flash attention: one wave = (head, 16-query tile), online softmax ----------
// 64-key tiles: 16 WMMAs per softmax butterfly + one LDS staging pass.
__global__ __launch_bounds__(256) void attn_flash_kernel(const __bf16* __restrict__ Q,
                                                         const __bf16* __restrict__ Kc,
                                                         const __bf16* __restrict__ VT,
                                                         __bf16* __restrict__ Out) {
  __shared__ __attribute__((aligned(32))) __bf16 plds[8][16 * 64];
  int lane = threadIdx.x & 31;
  int wave = threadIdx.x >> 5;
  int tile = blockIdx.x * 8 + wave;           // 16 heads * 256 qtiles
  int h = tile >> 8;
  int qt = (tile & 255) << 4;
  int l16 = lane & 15, half = lane >> 4;

  const __bf16* qrow = Q + (size_t)(qt + l16) * DIM_ + h * HDIM;
  v16bf aQ0 = *(const v16bf*)(qrow + half * 16);
  v16bf aQ1 = *(const v16bf*)(qrow + 32 + half * 16);

  v8f m_run = {-1e30f, -1e30f, -1e30f, -1e30f, -1e30f, -1e30f, -1e30f, -1e30f};
  v8f l_run = {0.f, 0.f, 0.f, 0.f, 0.f, 0.f, 0.f, 0.f};
  v8f Oa[4] = {};

  for (int kt = 0; kt < SEQ_; kt += 64) {
    // S tiles: 4 subtiles of 16 keys, contraction over head_dim (2 WMMAs each)
    v8f S[4];
#pragma unroll
    for (int j = 0; j < 4; j++) {
      const __bf16* kr = Kc + (size_t)(kt + j * 16 + l16) * DIM_ + h * HDIM + half * 16;
      v16bf b0 = *(const v16bf*)kr;
      v16bf b1 = *(const v16bf*)(kr + 32);
      v8f s = {};
      s = WMMA_BF16(aQ0, b0, s);
      s = WMMA_BF16(aQ1, b1, s);
      S[j] = s;
    }

    v8f alpha;
#pragma unroll
    for (int r = 0; r < 8; r++) {
      S[0][r] *= 0.125f;  // 1/sqrt(64)
      S[1][r] *= 0.125f;
      S[2][r] *= 0.125f;
      S[3][r] *= 0.125f;
      float t = fmaxf(fmaxf(S[0][r], S[1][r]), fmaxf(S[2][r], S[3][r]));
      t = red_max16(t);
      float mn = fmaxf(m_run[r], t);
      alpha[r] = __expf(m_run[r] - mn);
      S[0][r] = __expf(S[0][r] - mn);
      S[1][r] = __expf(S[1][r] - mn);
      S[2][r] = __expf(S[2][r] - mn);
      S[3][r] = __expf(S[3][r] - mn);
      float s = red_sum16((S[0][r] + S[1][r]) + (S[2][r] + S[3][r]));
      l_run[r] = l_run[r] * alpha[r] + s;
      m_run[r] = mn;
    }
#pragma unroll
    for (int r = 0; r < 8; r++) {
      Oa[0][r] *= alpha[r]; Oa[1][r] *= alpha[r];
      Oa[2][r] *= alpha[r]; Oa[3][r] *= alpha[r];
    }

    // P (D-layout) -> LDS row-major [16 q][64 k] -> reload as two A-fragments
    __bf16* pl = plds[wave];
#pragma unroll
    for (int r = 0; r < 8; r++) {
      int row = half * 8 + r;
      pl[row * 64 + l16]      = (__bf16)S[0][r];
      pl[row * 64 + 16 + l16] = (__bf16)S[1][r];
      pl[row * 64 + 32 + l16] = (__bf16)S[2][r];
      pl[row * 64 + 48 + l16] = (__bf16)S[3][r];
    }
    asm volatile("s_wait_dscnt 0" ::: "memory");
    const __bf16* pr = pl + l16 * 64 + half * 16;
    v16bf aP0 = *(const v16bf*)pr;
    v16bf aP1 = *(const v16bf*)(pr + 32);

#pragma unroll
    for (int j = 0; j < 4; j++) {
      const __bf16* vr = VT + (size_t)(h * HDIM + j * 16 + l16) * SEQ_ + kt + half * 16;
      v16bf bV0 = *(const v16bf*)vr;
      v16bf bV1 = *(const v16bf*)(vr + 32);
      Oa[j] = WMMA_BF16(aP0, bV0, Oa[j]);
      Oa[j] = WMMA_BF16(aP1, bV1, Oa[j]);
    }
  }

#pragma unroll
  for (int r = 0; r < 8; r++) {
    float inv = 1.f / l_run[r];
    int row = qt + half * 8 + r;
#pragma unroll
    for (int j = 0; j < 4; j++)
      Out[(size_t)row * DIM_ + h * HDIM + j * 16 + l16] = (__bf16)(Oa[j][r] * inv);
  }
}

// ---------- host orchestration ----------
extern "C" void kernel_launch(void* const* d_in, const int* in_sizes, int n_in,
                              void* d_out, int out_size, void* d_ws, size_t ws_size,
                              hipStream_t stream) {
  const float* x      = (const float*)d_in[0];
  const float* wq     = (const float*)d_in[1];
  const float* wk     = (const float*)d_in[2];
  const float* wv     = (const float*)d_in[3];
  const float* wo     = (const float*)d_in[4];
  const float* w1     = (const float*)d_in[5];
  const float* w2     = (const float*)d_in[6];
  const float* w3     = (const float*)d_in[7];
  const float* g_attn = (const float*)d_in[8];
  const float* g_ffn  = (const float*)d_in[9];

  char* ws = (char*)d_ws;
  const size_t MB = 1ull << 20;
  __bf16* wq_b = (__bf16*)(ws + 0 * MB);
  __bf16* wk_b = (__bf16*)(ws + 2 * MB);
  __bf16* wv_b = (__bf16*)(ws + 4 * MB);
  __bf16* wo_b = (__bf16*)(ws + 6 * MB);
  __bf16* w1_b = (__bf16*)(ws + 8 * MB);
  __bf16* w3_b = (__bf16*)(ws + 16 * MB);
  __bf16* w2_b = (__bf16*)(ws + 24 * MB);
  __bf16* xn_b = (__bf16*)(ws + 32 * MB);
  float*  q_f  = (float*)(ws + 40 * MB);
  float*  k_f  = (float*)(ws + 56 * MB);
  __bf16* q_b  = (__bf16*)(ws + 72 * MB);
  __bf16* k_b  = (__bf16*)(ws + 80 * MB);
  __bf16* vT_b = (__bf16*)(ws + 88 * MB);
  __bf16* at_b = (__bf16*)(ws + 96 * MB);
  float*  h_f  = (float*)(ws + 104 * MB);
  __bf16* hn_b = (__bf16*)(ws + 120 * MB);
  // FFN intermediates reuse freed regions (xn/q_f/k_f/q_b/k_b/vT_b dead by then)
  __bf16* a1_b = (__bf16*)(ws + 32 * MB);   // 32 MB, silu result written in place
  __bf16* a3_b = (__bf16*)(ws + 64 * MB);   // 32 MB

  const int nW = DIM_ * DIM_;       // 1M
  const int nH = HIDDEN_ * DIM_;    // 4M
  const int T256 = 256;

  // 1) weights -> bf16
  f32_to_bf16_kernel<<<(nW + 255) / 256, T256, 0, stream>>>(wq, wq_b, nW);
  f32_to_bf16_kernel<<<(nW + 255) / 256, T256, 0, stream>>>(wk, wk_b, nW);
  f32_to_bf16_kernel<<<(nW + 255) / 256, T256, 0, stream>>>(wv, wv_b, nW);
  f32_to_bf16_kernel<<<(nW + 255) / 256, T256, 0, stream>>>(wo, wo_b, nW);
  f32_to_bf16_kernel<<<(nH + 255) / 256, T256, 0, stream>>>(w1, w1_b, nH);
  f32_to_bf16_kernel<<<(nH + 255) / 256, T256, 0, stream>>>(w3, w3_b, nH);
  f32_to_bf16_kernel<<<(nH + 255) / 256, T256, 0, stream>>>(w2, w2_b, nH);

  // 2) rmsnorm(x) -> xn bf16
  rmsnorm_bf16_kernel<<<SEQ_, T256, 0, stream>>>(x, g_attn, xn_b, DIM_);

  // 3) QKV projections
  dim3 gProj(DIM_ / 256, SEQ_ / 64);
  gemm_wmma_kernel<0><<<gProj, T256, 0, stream>>>(xn_b, wq_b, q_f, nullptr, SEQ_, DIM_, DIM_);
  gemm_wmma_kernel<0><<<gProj, T256, 0, stream>>>(xn_b, wk_b, k_f, nullptr, SEQ_, DIM_, DIM_);
  gemm_wmma_kernel<2><<<gProj, T256, 0, stream>>>(xn_b, wv_b, vT_b, nullptr, SEQ_, DIM_, DIM_);

  // 4) RoPE -> bf16 q,k
  int nRope = SEQ_ * NHEADS * (HDIM / 2);
  rope_bf16_kernel<<<(nRope + 255) / 256, T256, 0, stream>>>(q_f, k_f, q_b, k_b, SEQ_);

  // 5) attention (16 heads * 256 qtiles = 4096 wave-tiles / 8 waves per block)
  attn_flash_kernel<<<(NHEADS * (SEQ_ / 16)) / 8, T256, 0, stream>>>(q_b, k_b, vT_b, at_b);

  // 6) h = x + attn @ wo^T
  gemm_wmma_kernel<0><<<gProj, T256, 0, stream>>>(at_b, wo_b, h_f, x, SEQ_, DIM_, DIM_);

  // 7) rmsnorm(h) -> hn bf16
  rmsnorm_bf16_kernel<<<SEQ_, T256, 0, stream>>>(h_f, g_ffn, hn_b, DIM_);

  // 8) FFN up projections (bf16 out)
  dim3 gUp(HIDDEN_ / 256, SEQ_ / 64);
  gemm_wmma_kernel<1><<<gUp, T256, 0, stream>>>(hn_b, w1_b, a1_b, nullptr, SEQ_, HIDDEN_, DIM_);
  gemm_wmma_kernel<1><<<gUp, T256, 0, stream>>>(hn_b, w3_b, a3_b, nullptr, SEQ_, HIDDEN_, DIM_);

  // 9) silu(a1)*a3 (in place into a1)
  int nFF = SEQ_ * HIDDEN_;
  silu_mul_bf16_kernel<<<(nFF + 255) / 256, T256, 0, stream>>>(a1_b, a3_b, nFF);

  // 10) out = h + ff @ w2^T  (f32 to d_out)
  dim3 gDown(DIM_ / 256, SEQ_ / 64);
  gemm_wmma_kernel<0><<<gDown, T256, 0, stream>>>(a1_b, w2_b, (float*)d_out, h_f, SEQ_, DIM_, HIDDEN_);
}
